// NeuromodulatedHolographicBrain_14413910245943
// MI455X (gfx1250) — matrix operand order repaired and checked by reading, not compiled, not run on hardware
//
#include <hip/hip_runtime.h>
#include <hip/hip_bf16.h>
#include <math.h>

// ---------------------------------------------------------------------------
// NeuromodulatedHolographicBrain step on MI455X (gfx1250).
// Densify the 1%-sparse COO matrices into bf16 (n-major) in the workspace
// (~84MB -> fits in 192MB L2), then run all matmuls as
// v_wmma_f32_16x16x32_bf16 dense GEMMs with fp32 accumulation.
// A-operand slabs are staged block-wide into LDS with
// global_load_async_to_lds_b128 (ASYNCcnt-tracked, triple-buffered) so the
// 8 waves of a block don't re-load the same 64 batch rows from global.
// Duplicate COO edges are summed with global_atomic_pk_add_bf16 (CDNA5).
// Workspace requirement: ~95 MB.
// ---------------------------------------------------------------------------

#define IN_SIZE  2048
#define HIDDEN   4096
#define BATCH    512
#define RBLOCKS  64
#define DT_CONST 0.1f

typedef unsigned short u16;
typedef unsigned int   u32;

typedef __attribute__((ext_vector_type(16))) __bf16 v16bf;
typedef __attribute__((ext_vector_type(8)))  float  v8f;
typedef __attribute__((ext_vector_type(4)))  u32    v4u;

// Workspace layout (bytes, all 256-aligned)
#define OFF_WT   0ull                       // [HIDDEN][IN_SIZE] bf16 (n-major W)
#define OFF_RT   16777216ull                // [HIDDEN][HIDDEN]  bf16 (n-major R)
#define OFF_PT   50331648ull                // [HIDDEN][HIDDEN]  bf16 (n-major P)
#define OFF_XB   83886080ull                // [BATCH][IN_SIZE]  bf16
#define OFF_HB   85983232ull                // [BATCH][HIDDEN]   bf16
#define OFF_HNB  90177536ull                // [BATCH][HIDDEN]   bf16 (h_new)
#define OFF_RWB  94371840ull                // [RBLOCKS][IN_SIZE] bf16
#define OFF_RG   94633984ull                // [BATCH][RBLOCKS]  f32
#define ZERO_V4U ((OFF_XB) / 16ull)         // dense W/R/P region, in 16B units

// LDS A-slab: 64 rows x 64 k bf16 per chunk, row stride padded to 72 u16
// (144B = 36 dwords -> 16 consecutive rows map to distinct bank groups).
#define AROW_PAD   72
#define ABUF_BYTES (64 * AROW_PAD * 2)      // 9216 B per buffer, 16B aligned

__device__ __forceinline__ u16 f32_to_bf16_rne(float f) {
  u32 u = __float_as_uint(f);
  u32 r = u + 0x7FFFu + ((u >> 16) & 1u);
  return (u16)(r >> 16);
}

union TileU { v16bf v; v4u q[2]; };

// A tile 16x32 bf16 from row-major src (ld elems/row), global path.
// ISA 16-bit A layout: m = lane&15; lanes<16 hold K {0..7,16..23}, lanes>=16
// hold K {8..15,24..31} -> two dwordx4 loads per lane.
__device__ __forceinline__ v16bf load_a_tile(const u16* __restrict__ src,
                                             int m0, int k0, int ld, int lane) {
  int m  = m0 + (lane & 15);
  int ko = k0 + ((lane >> 4) << 3);
  const u16* p = src + (size_t)m * ld + ko;
  TileU t;
  t.q[0] = *(const v4u*)(p);
  t.q[1] = *(const v4u*)(p + 16);
  return t.v;
}

// A tile 16x32 bf16 from the LDS slab (rows block-relative, padded stride).
__device__ __forceinline__ v16bf load_a_lds(const u16* sbuf, int msub, int kc,
                                            int lane) {
  int row = (msub << 4) + (lane & 15);
  int ko  = kc + ((lane >> 4) << 3);
  const u16* p = sbuf + row * AROW_PAD + ko;
  TileU t;
  t.q[0] = *(const v4u*)(p);        // ds_load_b128
  t.q[1] = *(const v4u*)(p + 16);
  return t.v;
}

// B tile 32x16 bf16 from n-major src: B[k][n] at src[n*ld + k].
// ISA 16-bit B layout: n = lane&15; lanes<16 hold K {0..7,16..23}, lanes>=16
// hold K {8..15,24..31} -> two dwordx4 loads per lane.
__device__ __forceinline__ v16bf load_b_tile(const u16* __restrict__ src,
                                             int n0, int k0, int ld, int lane) {
  int n  = n0 + (lane & 15);
  int ko = k0 + ((lane >> 4) << 3);
  const u16* p = src + (size_t)n * ld + ko;
  TileU t;
  t.q[0] = *(const v4u*)(p);
  t.q[1] = *(const v4u*)(p + 16);
  return t.v;
}

__device__ __forceinline__ v8f wmma_bf16(v16bf a, v16bf b, v8f c) {
  return __builtin_amdgcn_wmma_f32_16x16x32_bf16(false, a, false, b,
                                                 (short)0, c, false, false);
}

// Async-stage one 64x64 bf16 A chunk into LDS (whole block cooperates:
// 256 threads x 32B = 8KB payload, 2 x b128 async per thread).
// INST_OFFSET is added to BOTH the LDS and global address on this op.
__device__ __forceinline__ void stage_a_chunk(const u16* __restrict__ A, int ldA,
                                              int m0, int k0, u32 lds_off,
                                              int tid) {
  int row = tid >> 2;
  int c   = (tid & 3) << 4;                       // u16 elems within row
  const u16* g = A + (size_t)(m0 + row) * ldA + k0 + c;
  u32 l = lds_off + row * (AROW_PAD * 2) + (c << 1);
  asm volatile("global_load_async_to_lds_b128 %0, %1, off"
               :: "v"(l), "v"(g) : "memory");
  asm volatile("global_load_async_to_lds_b128 %0, %1, off offset:16"
               :: "v"(l), "v"(g) : "memory");
}

// One GEMM phase: acc[0..3] (64x16 per wave) += A[m0:m0+64, 0:K] @ B.
// A staged through LDS (triple-buffered async), B streamed from global.
__device__ __forceinline__ void mm_phase(v8f* acc,
                                         const u16* __restrict__ A, int ldA,
                                         const u16* __restrict__ B, int ldB,
                                         int K, int m0, int n0,
                                         const u16* sA, u32 lds_base,
                                         int lane, int tid) {
  __syncthreads();                                  // protect prior buffer users
  stage_a_chunk(A, ldA, m0, 0, lds_base, tid);
  int nch = K >> 6;
  for (int ch = 0; ch < nch; ++ch) {
    int buf = ch % 3;
    if (ch + 1 < nch) {
      stage_a_chunk(A, ldA, m0, (ch + 1) << 6,
                    lds_base + ((ch + 1) % 3) * ABUF_BYTES, tid);
      asm volatile("s_wait_asynccnt 0x2" ::: "memory");   // chunk ch landed
    } else {
      asm volatile("s_wait_asynccnt 0x0" ::: "memory");
    }
    __syncthreads();
    const u16* sbuf = sA + buf * (64 * AROW_PAD);
    int kg = ch << 6;
#pragma unroll
    for (int kc = 0; kc < 64; kc += 32) {
      const u16* bp = B + (size_t)(n0 + (lane & 15)) * ldB + kg + kc;
      __builtin_prefetch(bp + 128, 0, 0);           // global_prefetch B stream
      v16bf b = load_b_tile(B, n0, kg + kc, ldB, lane);
#pragma unroll
      for (int s = 0; s < 4; ++s) {
        v16bf a = load_a_lds(sbuf, s, kc, lane);
        acc[s] = wmma_bf16(a, b, acc[s]);
      }
    }
  }
}

// --------------------------- prep: zero + cvt ------------------------------
__global__ void prep_kernel(const float* __restrict__ x,
                            const float* __restrict__ h,
                            const float* __restrict__ rw,
                            v4u* __restrict__ zero_base,
                            u32* __restrict__ xb, u32* __restrict__ hb,
                            u32* __restrict__ rwb) {
  long tid = blockIdx.x * (long)blockDim.x + threadIdx.x;
  long nth = gridDim.x * (long)blockDim.x;
  v4u z = (v4u){0u, 0u, 0u, 0u};
  for (long i = tid; i < (long)ZERO_V4U; i += nth) zero_base[i] = z;
  for (long i = tid; i < (long)BATCH * IN_SIZE / 2; i += nth) {
    u32 lo = f32_to_bf16_rne(x[2 * i]);
    u32 hi = f32_to_bf16_rne(x[2 * i + 1]);
    xb[i] = lo | (hi << 16);
  }
  for (long i = tid; i < (long)BATCH * HIDDEN / 2; i += nth) {
    u32 lo = f32_to_bf16_rne(h[2 * i]);
    u32 hi = f32_to_bf16_rne(h[2 * i + 1]);
    hb[i] = lo | (hi << 16);
  }
  for (long i = tid; i < (long)RBLOCKS * IN_SIZE / 2; i += nth) {
    u32 lo = f32_to_bf16_rne(rw[2 * i]);
    u32 hi = f32_to_bf16_rne(rw[2 * i + 1]);
    rwb[i] = lo | (hi << 16);
  }
}

// ------------------- COO scatter-add into dense bf16 -----------------------
// dst is n-major: element [k=rows[e]][n=cols[e]] at dst[cols[e]*K + rows[e]].
// Duplicates must SUM (reference scatter-add) -> global_atomic_pk_add_bf16.
__global__ void scatter_kernel(const int* __restrict__ rows,
                               const int* __restrict__ cols,
                               const float* __restrict__ vals,
                               u16* __restrict__ dst, int K, int nnz) {
  int e = blockIdx.x * blockDim.x + threadIdx.x;
  if (e >= nnz) return;
  long idx = (long)cols[e] * K + rows[e];
  u32 bf = f32_to_bf16_rne(vals[e]);
  u32 packed = (idx & 1) ? (bf << 16) : bf;   // other half adds +0.0 (no-op)
  u16* p = dst + (idx & ~1L);                 // 4B-aligned bf16 pair
  asm volatile("global_atomic_pk_add_bf16 %0, %1, off"
               :: "v"(p), "v"(packed) : "memory");
}

// --------------------------- router GEMM -----------------------------------
// r_gate[512,64] = sigmoid(x @ router_w^T + b). 32 waves, each 64x16 tile.
__global__ void router_kernel(const u16* __restrict__ xb,
                              const u16* __restrict__ rwb,
                              const float* __restrict__ rbias,
                              float* __restrict__ rg) {
  int lane = threadIdx.x & 31;
  int wave = blockIdx.x * (blockDim.x >> 5) + (threadIdx.x >> 5);
  int m0 = (wave >> 2) * 64;
  int n0 = (wave & 3) * 16;
  v8f acc[4];
#pragma unroll
  for (int s = 0; s < 4; ++s) acc[s] = (v8f)0.0f;

  for (int k0 = 0; k0 < IN_SIZE; k0 += 32) {
    v16bf b = load_b_tile(rwb, n0, k0, IN_SIZE, lane);
#pragma unroll
    for (int s = 0; s < 4; ++s) {
      v16bf a = load_a_tile(xb, m0 + 16 * s, k0, IN_SIZE, lane);
      acc[s] = wmma_bf16(a, b, acc[s]);
    }
  }
  int nl = lane & 15, hi = lane >> 4;
  int n = n0 + nl;
  float bv = rbias[n];
#pragma unroll
  for (int s = 0; s < 4; ++s)
#pragma unroll
    for (int r = 0; r < 8; ++r) {
      int m = m0 + 16 * s + r + 8 * hi;
      float v = acc[s][r] + bv;
      rg[(size_t)m * RBLOCKS + n] = 1.0f / (1.0f + expf(-v));
    }
}

// ----------------- fused sensory + recurrent + update ----------------------
// acc = (x@W + Wb) * mask ; acc += h@R ; t = tanh(acc + Rb)
// h_new = h_prev + gate*(t - h_prev)*(DT/tau)
// Block: 8 waves = 64 rows x 128 cols. grid = (32, 8).
__global__ void gemm_wr_kernel(const u16* __restrict__ xb,
                               const u16* __restrict__ hb,
                               const u16* __restrict__ wt,
                               const u16* __restrict__ rt,
                               const float* __restrict__ wbias,
                               const float* __restrict__ rbias,
                               const float* __restrict__ rg,
                               const float* __restrict__ h_prev,
                               const float* __restrict__ gate,
                               const float* __restrict__ tau,
                               float* __restrict__ h_new,
                               u16* __restrict__ hnew_bf) {
  __shared__ __attribute__((aligned(16))) u16 sA[3][64][AROW_PAD];
  int tid = threadIdx.x;
  int lane = tid & 31;
  int w = tid >> 5;
  int m0 = blockIdx.y * 64;
  int n0 = blockIdx.x * 128 + w * 16;
  u32 lds_base = (u32)(size_t)(&sA[0][0][0]);
  v8f acc[4];
#pragma unroll
  for (int s = 0; s < 4; ++s) acc[s] = (v8f)0.0f;

  // phase 1: x @ W  (K = IN_SIZE)
  mm_phase(acc, xb, IN_SIZE, wt, IN_SIZE, IN_SIZE, m0, n0,
           &sA[0][0][0], lds_base, lane, tid);

  int nl = lane & 15, hi = lane >> 4;
  int n = n0 + nl;
  int rb = n >> 6;  // router block = n / REPEAT(64); constant per wave
  float wb = wbias[n];
#pragma unroll
  for (int s = 0; s < 4; ++s)
#pragma unroll
    for (int r = 0; r < 8; ++r) {
      int m = m0 + 16 * s + r + 8 * hi;
      acc[s][r] = (acc[s][r] + wb) * rg[(size_t)m * RBLOCKS + rb];
    }

  // phase 2: + h_prev @ R  (K = HIDDEN), WMMA chains C accumulation
  mm_phase(acc, hb, HIDDEN, rt, HIDDEN, HIDDEN, m0, n0,
           &sA[0][0][0], lds_base, lane, tid);

  float rbv = rbias[n];
  float dt_tau = DT_CONST / tau[n];
#pragma unroll
  for (int s = 0; s < 4; ++s)
#pragma unroll
    for (int r = 0; r < 8; ++r) {
      int m = m0 + 16 * s + r + 8 * hi;
      float t = tanhf(acc[s][r] + rbv);
      float hp = h_prev[(size_t)m * HIDDEN + n];
      float hn = hp + gate[m] * (t - hp) * dt_tau;
      h_new[(size_t)m * HIDDEN + n] = hn;
      hnew_bf[(size_t)m * HIDDEN + n] = f32_to_bf16_rne(hn);
    }
}

// --------------------------- pred = h_new @ P + b --------------------------
__global__ void gemm_p_kernel(const u16* __restrict__ hnb,
                              const u16* __restrict__ pt,
                              const float* __restrict__ pbias,
                              float* __restrict__ pred) {
  __shared__ __attribute__((aligned(16))) u16 sA[3][64][AROW_PAD];
  int tid = threadIdx.x;
  int lane = tid & 31;
  int w = tid >> 5;
  int m0 = blockIdx.y * 64;
  int n0 = blockIdx.x * 128 + w * 16;
  u32 lds_base = (u32)(size_t)(&sA[0][0][0]);
  v8f acc[4];
#pragma unroll
  for (int s = 0; s < 4; ++s) acc[s] = (v8f)0.0f;

  mm_phase(acc, hnb, HIDDEN, pt, HIDDEN, HIDDEN, m0, n0,
           &sA[0][0][0], lds_base, lane, tid);

  int nl = lane & 15, hi = lane >> 4;
  int n = n0 + nl;
  float bv = pbias[n];
#pragma unroll
  for (int s = 0; s < 4; ++s)
#pragma unroll
    for (int r = 0; r < 8; ++r) {
      int m = m0 + 16 * s + r + 8 * hi;
      pred[(size_t)m * HIDDEN + n] = acc[s][r] + bv;
    }
}

// ---------------------------------------------------------------------------
extern "C" void kernel_launch(void* const* d_in, const int* in_sizes, int n_in,
                              void* d_out, int out_size, void* d_ws, size_t ws_size,
                              hipStream_t stream) {
  (void)n_in; (void)out_size; (void)ws_size;  // needs ~95 MB workspace
  const float* x        = (const float*)d_in[0];
  const float* h_prev   = (const float*)d_in[1];
  const float* gate     = (const float*)d_in[2];
  const float* W_vals   = (const float*)d_in[3];
  const float* W_bias   = (const float*)d_in[4];
  const float* R_vals   = (const float*)d_in[5];
  const float* R_bias   = (const float*)d_in[6];
  const float* P_vals   = (const float*)d_in[7];
  const float* P_bias   = (const float*)d_in[8];
  const float* router_w = (const float*)d_in[9];
  const float* router_b = (const float*)d_in[10];
  const float* tau      = (const float*)d_in[11];
  const int* W_rows = (const int*)d_in[12];
  const int* W_cols = (const int*)d_in[13];
  const int* R_rows = (const int*)d_in[14];
  const int* R_cols = (const int*)d_in[15];
  const int* P_rows = (const int*)d_in[16];
  const int* P_cols = (const int*)d_in[17];
  int nnz_w = in_sizes[3], nnz_r = in_sizes[5], nnz_p = in_sizes[7];

  char* ws = (char*)d_ws;
  u16* wt  = (u16*)(ws + OFF_WT);
  u16* rt  = (u16*)(ws + OFF_RT);
  u16* pt  = (u16*)(ws + OFF_PT);
  u16* xb  = (u16*)(ws + OFF_XB);
  u16* hb  = (u16*)(ws + OFF_HB);
  u16* hnb = (u16*)(ws + OFF_HNB);
  u16* rwb = (u16*)(ws + OFF_RWB);
  float* rg = (float*)(ws + OFF_RG);

  float* h_new = (float*)d_out;
  float* pred  = (float*)d_out + (size_t)BATCH * HIDDEN;

  prep_kernel<<<2048, 256, 0, stream>>>(x, h_prev, router_w, (v4u*)ws,
                                        (u32*)xb, (u32*)hb, (u32*)rwb);
  scatter_kernel<<<(nnz_w + 255) / 256, 256, 0, stream>>>(W_rows, W_cols, W_vals,
                                                          wt, IN_SIZE, nnz_w);
  scatter_kernel<<<(nnz_r + 255) / 256, 256, 0, stream>>>(R_rows, R_cols, R_vals,
                                                          rt, HIDDEN, nnz_r);
  scatter_kernel<<<(nnz_p + 255) / 256, 256, 0, stream>>>(P_rows, P_cols, P_vals,
                                                          pt, HIDDEN, nnz_p);
  router_kernel<<<4, 256, 0, stream>>>(xb, rwb, router_b, rg);

  dim3 g(32, 8);
  gemm_wr_kernel<<<g, 256, 0, stream>>>(xb, hb, wt, rt, W_bias, R_bias, rg,
                                        h_prev, gate, tau, h_new, hnb);
  gemm_p_kernel<<<g, 256, 0, stream>>>(hnb, pt, P_bias, pred);
}
